// Kmeans_67370857005154
// MI455X (gfx1250) — matrix operand order repaired and checked by reading, not compiled, not run on hardware
//
#include <hip/hip_runtime.h>
#include <hip/hip_bf16.h>

// K-means for MI455X (gfx1250).
//  - x converted once to bf16 (102.4 MB -> L2-resident on the 192MB L2).
//  - Assignment matmul: V_WMMA_F32_16X16X32_BF16, 8 accumulators/wave (32 rows x 64 centroids).
//  - Centroids staged block-wide into LDS with GLOBAL_LOAD_ASYNC_TO_LDS_B128 (+ s_wait_asynccnt).
//  - Flattened 64-step MAC loop: B-fragments prefetched one k-step ahead (LOADcnt),
//    A-fragments prefetched one step ahead from LDS (DScnt) -> partial waits, no dead waits.
//  - In-register argmin + __shfl_xor(16) merge; no score LDS, no extra barrier.
//
// Workspace (~102.7 MB): [xb: N*D bf16][cent: 64*512 bf16][c_sq: 64 f32][sums: 64*512 f32][counts: 64 f32]

typedef __attribute__((ext_vector_type(16))) __bf16 bf16x16;
typedef __attribute__((ext_vector_type(8)))  __bf16 bf16x8;
typedef __attribute__((ext_vector_type(4)))  __bf16 bf16x4;
typedef __attribute__((ext_vector_type(8)))  float  f32x8;

#define DDIM  512
#define KC    64
#define NITER 10

// ---------------- fp32 -> bf16 conversion of x ----------------
__global__ __launch_bounds__(256) void convert_kernel(const float* __restrict__ x,
                                                      __bf16* __restrict__ xb, size_t n) {
  size_t i = ((size_t)blockIdx.x * 256 + threadIdx.x) * 4;
  if (i < n) {
    float4 v = *(const float4*)(x + i);
    bf16x4 o;
    o[0] = (__bf16)v.x; o[1] = (__bf16)v.y; o[2] = (__bf16)v.z; o[3] = (__bf16)v.w;
    *(bf16x4*)(xb + i) = o;
  }
}

// ---------------- centroid init (strided seed rows) ----------------
__global__ __launch_bounds__(256) void init_cent_kernel(const float* __restrict__ x,
                                                        __bf16* __restrict__ cent,
                                                        float* __restrict__ c_sq, int N) {
  __shared__ float red[256];
  int k = blockIdx.x, tid = threadIdx.x;
  const float* src = x + (size_t)(k * (N / KC)) * DDIM;
  float ss = 0.f;
  for (int d = tid; d < DDIM; d += 256) {
    float c = src[d];
    cent[(size_t)k * DDIM + d] = (__bf16)c;
    ss += c * c;
  }
  red[tid] = ss; __syncthreads();
  for (int o = 128; o > 0; o >>= 1) { if (tid < o) red[tid] += red[tid + o]; __syncthreads(); }
  if (tid == 0) c_sq[k] = red[0];
}

// ---------------- WMMA fragment loads ----------------
// A 16x32 bf16 (from LDS): lane holds row M=lane%16; elems 0..7 = K[8*half..], elems 8..15 = K[16+8*half..]
__device__ __forceinline__ bf16x16 load_a_frag(const __bf16* row, int kk, int half) {
  bf16x8 lo = *(const bf16x8*)(row + kk + half * 8);
  bf16x8 hi = *(const bf16x8*)(row + kk + 16 + half * 8);
  return __builtin_shufflevector(lo, hi, 0,1,2,3,4,5,6,7,8,9,10,11,12,13,14,15);
}
// B 32x16 bf16 (from global): lane holds col N=lane%16; elems 0..7 = K[8*half..], elems 8..15 = K[16+8*half..]
__device__ __forceinline__ bf16x16 load_b_frag(const __bf16* row, int kk, int half) {
  bf16x8 lo = *(const bf16x8*)(row + kk + half * 8);
  bf16x8 hi = *(const bf16x8*)(row + kk + 16 + half * 8);
  return __builtin_shufflevector(lo, hi, 0,1,2,3,4,5,6,7,8,9,10,11,12,13,14,15);
}

// ---------------- assignment: argmin_k ( c_sq[k] - 2 * <c_k, x_n> ) ----------------
__global__ __launch_bounds__(256, 1) void assign_kernel(const __bf16* __restrict__ xb,
                                                        const __bf16* __restrict__ cent,
                                                        const float* __restrict__ c_sq,
                                                        int* __restrict__ out, int N) {
  __shared__ __bf16 s_cent[KC * DDIM];   // 64 KB centroid stage
  int tid  = threadIdx.x;
  int wave = tid >> 5;
  int lane = tid & 31;
  int half = lane >> 4;
  int ln   = lane & 15;

  // ---- async-stage all centroids into LDS: 4096 x 16B, 16 chunks/thread ----
  {
#pragma unroll
    for (int i = 0; i < (KC * DDIM * 2 / 16) / 256; ++i) {
      int idx = tid + i * 256;                                   // 16B chunk index
      unsigned lofs = (unsigned)(uintptr_t)(s_cent + idx * 8);   // low 32 bits = LDS offset
      unsigned long long ga = (unsigned long long)(uintptr_t)(cent + idx * 8);
      asm volatile("global_load_async_to_lds_b128 %0, %1, off"
                   :: "v"(lofs), "v"(ga) : "memory");
    }
    asm volatile("s_wait_asynccnt 0x0" ::: "memory");
    __syncthreads();
  }

  // ---- each wave: 32 x-rows (two 16-col B tiles) vs all 64 centroids ----
  long rowbase = (long)blockIdx.x * 256 + wave * 32;
  long xr0 = rowbase + ln;       if (xr0 > (long)N - 1) xr0 = (long)N - 1;  // clamp: keep EXEC full
  long xr1 = rowbase + 16 + ln;  if (xr1 > (long)N - 1) xr1 = (long)N - 1;
  const __bf16* xrow0 = xb + (size_t)xr0 * DDIM;
  const __bf16* xrow1 = xb + (size_t)xr1 * DDIM;
  const __bf16* crow[4];
#pragma unroll
  for (int t = 0; t < 4; ++t) crow[t] = s_cent + (size_t)(t * 16 + ln) * DDIM;

  f32x8 acc0[4] = {}, acc1[4] = {};
  bf16x16 b0  = load_b_frag(xrow0, 0, half);
  bf16x16 b1  = load_b_frag(xrow1, 0, half);
  bf16x16 nb0 = b0, nb1 = b1;
  bf16x16 a   = load_a_frag(crow[0], 0, half);

  // flattened 64-step MAC loop: q = s*4 + t; A prefetched 1 step ahead, B 1 k-step ahead
#pragma unroll
  for (int q = 0; q < (DDIM / 32) * 4; ++q) {
    int s = q >> 2, t = q & 3;
    bf16x16 na = a;
    if (q < (DDIM / 32) * 4 - 1) {              // compile-time under full unroll
      int qn = q + 1;
      na = load_a_frag(crow[qn & 3], (qn >> 2) * 32, half);   // ds_load_b128 x2, in flight during WMMAs
    }
    if (t == 0 && s < DDIM / 32 - 1) {
      nb0 = load_b_frag(xrow0, (s + 1) * 32, half);
      nb1 = load_b_frag(xrow1, (s + 1) * 32, half);
    }
    acc0[t] = __builtin_amdgcn_wmma_f32_16x16x32_bf16(false, a, false, b0, (short)0, acc0[t], false, false);
    acc1[t] = __builtin_amdgcn_wmma_f32_16x16x32_bf16(false, a, false, b1, (short)0, acc1[t], false, false);
    a = na;
    if (t == 3) { b0 = nb0; b1 = nb1; }
  }

  // ---- in-register argmin. C layout: lane l, vgpr r -> M = r + 8*(l/16), N = l%16 ----
#pragma unroll
  for (int j = 0; j < 2; ++j) {
    float best = 3.4e38f; int bi = 0;
#pragma unroll
    for (int t = 0; t < 4; ++t) {
#pragma unroll
      for (int r = 0; r < 8; ++r) {
        int m = t * 16 + 8 * half + r;          // ascending per lane -> '<' keeps smallest k
        float dsc = c_sq[m] - 2.0f * (j ? acc1[t][r] : acc0[t][r]);
        if (dsc < best) { best = dsc; bi = m; }
      }
    }
    // merge lane l with lane l^16 (other half of the M range), tie-break on smaller index
    float obest = __shfl_xor(best, 16, 32);
    int   obi   = __shfl_xor(bi, 16, 32);
    if (obest < best || (obest == best && obi < bi)) { best = obest; bi = obi; }
    long row = rowbase + j * 16 + ln;
    if (half == 0 && row < N) out[row] = bi;
  }
}

// ---------------- zero sums+counts ----------------
__global__ __launch_bounds__(256) void zero_kernel(float* __restrict__ p, int n) {
  int i = blockIdx.x * 256 + threadIdx.x;
  if (i < n) p[i] = 0.f;
}

// ---------------- segment-sum update (LDS staged) ----------------
__global__ __launch_bounds__(256) void accum_kernel(const float* __restrict__ x,
                                                    const int* __restrict__ assign,
                                                    float* __restrict__ sums,
                                                    float* __restrict__ counts, int N) {
  __shared__ float s_sums[KC][128];   // 32 KB slice of the 64x512 accumulator
  __shared__ float s_cnt[KC];
  int tid   = threadIdx.x;
  int dbase = blockIdx.x * 128;       // 4 D-slices
  int rowbeg = blockIdx.y * 1024;
  int rowend = rowbeg + 1024; if (rowend > N) rowend = N;

  for (int i = tid; i < KC * 128; i += 256) ((float*)s_sums)[i] = 0.f;
  if (tid < KC) s_cnt[tid] = 0.f;
  __syncthreads();

  int col  = tid & 127;
  int half = tid >> 7;
  bool docnt = (blockIdx.x == 0) && (col == 0);
  for (int r = rowbeg + half; r < rowend; r += 2) {
    int k = assign[r];
    atomicAdd(&s_sums[k][col], x[(size_t)r * DDIM + dbase + col]);   // ds_add_f32
    if (docnt) atomicAdd(&s_cnt[k], 1.0f);
  }
  __syncthreads();

  for (int i = tid; i < KC * 128; i += 256) {
    int k = i >> 7, d = i & 127;
    float v = s_sums[k][d];
    if (v != 0.f) atomicAdd(&sums[(size_t)k * DDIM + dbase + d], v); // global_atomic_add_f32
  }
  if (blockIdx.x == 0 && tid < KC && s_cnt[tid] != 0.f)
    atomicAdd(&counts[tid], s_cnt[tid]);
}

// ---------------- finalize: centroids = sums/counts (bf16) + c_sq ----------------
__global__ __launch_bounds__(256) void finalize_kernel(const float* __restrict__ sums,
                                                       const float* __restrict__ counts,
                                                       __bf16* __restrict__ cent,
                                                       float* __restrict__ c_sq) {
  __shared__ float red[256];
  int k = blockIdx.x, tid = threadIdx.x;
  float cnt = counts[k];
  float ss = 0.f;
  for (int d = tid; d < DDIM; d += 256) {
    float c = sums[(size_t)k * DDIM + d] / cnt;   // 0/0 -> NaN, matches reference semantics
    cent[(size_t)k * DDIM + d] = (__bf16)c;
    ss += c * c;
  }
  red[tid] = ss; __syncthreads();
  for (int o = 128; o > 0; o >>= 1) { if (tid < o) red[tid] += red[tid + o]; __syncthreads(); }
  if (tid == 0) c_sq[k] = red[0];
}

extern "C" void kernel_launch(void* const* d_in, const int* in_sizes, int n_in,
                              void* d_out, int out_size, void* d_ws, size_t ws_size,
                              hipStream_t stream) {
  const float* x = (const float*)d_in[0];
  int N = in_sizes[0] / DDIM;

  char* ws = (char*)d_ws;
  size_t nb_xb = (size_t)N * DDIM * sizeof(__bf16);
  __bf16* xb = (__bf16*)ws;
  size_t off = (nb_xb + 255) & ~(size_t)255;
  __bf16* cent = (__bf16*)(ws + off); off += (size_t)KC * DDIM * sizeof(__bf16);
  float* c_sq  = (float*)(ws + off);  off += 256;
  float* sums  = (float*)(ws + off);  off += (size_t)KC * DDIM * sizeof(float);
  float* counts = (float*)(ws + off); off += 256;   // contiguous after sums

  int* assign = (int*)d_out;   // int32 assignments; final iteration's contents are the output

  size_t nelem = (size_t)N * DDIM;
  convert_kernel<<<(unsigned)((nelem / 4 + 255) / 256), 256, 0, stream>>>(x, xb, nelem);
  init_cent_kernel<<<KC, 256, 0, stream>>>(x, cent, c_sq, N);

  int ablocks = (N + 255) / 256;
  int chunks  = (N + 1023) / 1024;
  for (int t = 0; t <= NITER; ++t) {
    assign_kernel<<<ablocks, 256, 0, stream>>>(xb, cent, c_sq, assign, N);
    if (t < NITER) {
      zero_kernel<<<(KC * DDIM + KC + 255) / 256, 256, 0, stream>>>(sums, KC * DDIM + KC);
      accum_kernel<<<dim3(DDIM / 128, chunks), 256, 0, stream>>>(x, assign, sums, counts, N);
      finalize_kernel<<<KC, 256, 0, stream>>>(sums, counts, cent, c_sq);
    }
  }
}
